// MahalanobisKDE_74259984548530
// MI455X (gfx1250) — matrix operand order repaired and checked by reading, not compiled, not run on hardware
//
#include <hip/hip_runtime.h>

// Problem constants (match reference)
constexpr int Bdim = 4096;
constexpr int Ndim = 16384;
constexpr int Ddim = 1024;

// Tiling
constexpr int BM = 128;     // rows of X per block
constexpr int BN = 128;     // svs rows per tile
constexpr int BK = 32;      // K per WMMA step (bf16 16x16x32)
constexpr int KSTEPS = Ddim / BK;
constexpr int LDST = 40;    // LDS row stride in halves (32 + 8 pad -> 16B aligned, conflict-free)
constexpr int NSPLIT = 32;  // N-dimension splits (partial logsumexp)

typedef __attribute__((ext_vector_type(16))) __bf16 v16bf;
typedef __attribute__((ext_vector_type(8)))  float  v8f;

struct Q32 { uint4 a, b; };

__device__ __forceinline__ v16bf mkfrag(const unsigned short* p0, const unsigned short* p1) {
    Q32 q;
    q.a = *(const uint4*)p0;   // ds_load_b128
    q.b = *(const uint4*)p1;   // ds_load_b128
    return __builtin_bit_cast(v16bf, q);
}

__device__ __forceinline__ v8f wmma_bf16(v16bf a, v16bf b, v8f c) {
    // D = A(16x32 bf16) * B(32x16 bf16) + C(16x16 f32)
    return __builtin_amdgcn_wmma_f32_16x16x32_bf16(false, a, false, b, (short)0, c, false, false);
}

// Async global -> LDS copy, 128 bits per lane. Tracked by ASYNCcnt.
// VDST VGPR carries the LDS byte offset (generic LDS pointers keep the LDS
// offset in addr[31:0] per CDNA5 ISA 10.2 aperture decode).
__device__ __forceinline__ void async_b128(const unsigned short* g, unsigned short* l) {
    asm volatile("global_load_async_to_lds_b128 %0, %1, off"
                 :: "v"((unsigned)(unsigned long long)l),
                    "v"((unsigned long long)g)
                 : "memory");
}
__device__ __forceinline__ void wait_async0() {
    asm volatile("s_wait_asynccnt 0x0" ::: "memory");
}

__device__ __forceinline__ unsigned short f2bf(float f) {
    unsigned u = __float_as_uint(f);
    u += 0x7FFFu + ((u >> 16) & 1u);   // round to nearest even
    return (unsigned short)(u >> 16);
}
__device__ __forceinline__ float bf2f(unsigned short h) {
    return __uint_as_float(((unsigned)h) << 16);
}

// ---------------------------------------------------------------------------
// Kernel 1: fp32 row -> (bf16 hi, bf16 lo) split + exact f32 squared norm.
// ---------------------------------------------------------------------------
__global__ __launch_bounds__(256) void kde_split(const float* __restrict__ src,
                                                 unsigned short* __restrict__ hi,
                                                 unsigned short* __restrict__ lo,
                                                 float* __restrict__ nrm) {
    const int row = blockIdx.x;
    const int tid = threadIdx.x;
    const float* r = src + (size_t)row * Ddim;
    float ss = 0.f;
    for (int j = tid; j < Ddim; j += 256) {
        float x = r[j];
        ss += x * x;
        unsigned short h = f2bf(x);
        float fh = bf2f(h);
        unsigned short l = f2bf(x - fh);
        hi[(size_t)row * Ddim + j] = h;
        lo[(size_t)row * Ddim + j] = l;
    }
    __shared__ float red[256];
    red[tid] = ss;
    __syncthreads();
    for (int s = 128; s > 0; s >>= 1) {
        if (tid < s) red[tid] += red[tid + s];
        __syncthreads();
    }
    if (tid == 0) nrm[row] = red[0];
}

// ---------------------------------------------------------------------------
// Kernel 2: fused bf16x3 WMMA GEMM (X . svs^T) + squared-distance epilogue +
// online per-row logsumexp. Double-buffered LDS staging via async-to-LDS.
// Block: 256 threads = 8 waves (2 M-waves x 4 N-waves); each wave owns 4x2
// 16x16 subtiles -> 24 v_wmma_f32_16x16x32_bf16 per K-step.
// ---------------------------------------------------------------------------
__global__ __launch_bounds__(256) void kde_gemm_lse(
        const unsigned short* __restrict__ Xhi, const unsigned short* __restrict__ Xlo,
        const unsigned short* __restrict__ Shi, const unsigned short* __restrict__ Slo,
        const float* __restrict__ x2, const float* __restrict__ s2,
        const float* __restrict__ gptr,
        float* __restrict__ partM, float* __restrict__ partL) {
    __shared__ unsigned short tXh[2][BM * LDST];
    __shared__ unsigned short tXl[2][BM * LDST];
    __shared__ unsigned short tSh[2][BN * LDST];
    __shared__ unsigned short tSl[2][BN * LDST];
    __shared__ float    sx2[BM];
    __shared__ unsigned rowMu[BM];   // float bits; values <= 0 so unsigned-min == float-max
    __shared__ float    rowL[BM];

    const int tid   = threadIdx.x;
    const int m0    = blockIdx.x * BM;
    const int split = blockIdx.y;
    const int n0    = split * (Ndim / NSPLIT);
    const float s   = -gptr[0];      // lse scale (= -gamma)

    if (tid < BM) {
        sx2[tid]   = x2[m0 + tid];
        rowMu[tid] = 0xFF800000u;    // bits(-inf)
        rowL[tid]  = 0.f;
    }

    const int w    = tid >> 5;
    const int lane = tid & 31;
    const int wm   = w & 1;          // M half: rows wm*64..
    const int wn   = w >> 1;         // N quarter: cols wn*32..
    const int lr   = lane & 15;
    const int lh   = lane >> 4;

    const int ldrow = tid >> 1;           // 0..127: tile row this thread stages
    const int ldc   = (tid & 1) << 4;     // half-offset 0 or 16 in the 32-wide K slab
    const int lofs  = ldrow * LDST + ldc;

    const unsigned short* gXh = Xhi + (size_t)(m0 + ldrow) * Ddim + ldc;
    const unsigned short* gXl = Xlo + (size_t)(m0 + ldrow) * Ddim + ldc;

    float rm[4][8], rl[4][8];
#pragma unroll
    for (int i = 0; i < 4; i++)
#pragma unroll
        for (int j = 0; j < 8; j++) { rm[i][j] = -__builtin_inff(); rl[i][j] = 0.f; }

    for (int nt = 0; nt < Ndim / NSPLIT; nt += BN) {
        const int nb = n0 + nt;
        const unsigned short* gSh = Shi + (size_t)(nb + ldrow) * Ddim + ldc;
        const unsigned short* gSl = Slo + (size_t)(nb + ldrow) * Ddim + ldc;

        v8f acc[4][2];
#pragma unroll
        for (int i = 0; i < 4; i++)
#pragma unroll
            for (int j = 0; j < 2; j++)
#pragma unroll
                for (int k = 0; k < 8; k++) acc[i][j][k] = 0.f;

        // Stage K-slab 0 into buffer 0 (async, no VGPR transit).
        {
            async_b128(gXh,     &tXh[0][lofs]);  async_b128(gXh + 8, &tXh[0][lofs + 8]);
            async_b128(gXl,     &tXl[0][lofs]);  async_b128(gXl + 8, &tXl[0][lofs + 8]);
            async_b128(gSh,     &tSh[0][lofs]);  async_b128(gSh + 8, &tSh[0][lofs + 8]);
            async_b128(gSl,     &tSl[0][lofs]);  async_b128(gSl + 8, &tSl[0][lofs + 8]);
        }
        wait_async0();
        __syncthreads();

        for (int ks = 0; ks < KSTEPS; ks++) {
            const int cur = ks & 1;
            // Issue async loads for the next K-slab before computing this one:
            // their latency hides under the 24 WMMAs below.
            if (ks + 1 < KSTEPS) {
                const int nxt = cur ^ 1;
                const int ko  = (ks + 1) * BK;
                async_b128(gXh + ko,     &tXh[nxt][lofs]);
                async_b128(gXh + ko + 8, &tXh[nxt][lofs + 8]);
                async_b128(gXl + ko,     &tXl[nxt][lofs]);
                async_b128(gXl + ko + 8, &tXl[nxt][lofs + 8]);
                async_b128(gSh + ko,     &tSh[nxt][lofs]);
                async_b128(gSh + ko + 8, &tSh[nxt][lofs + 8]);
                async_b128(gSl + ko,     &tSl[nxt][lofs]);
                async_b128(gSl + ko + 8, &tSl[nxt][lofs + 8]);
            }

            // A frags: lane = row; halves [8*lh..+7] = K(8lh..), then [16+8lh..]
            v16bf ah[4], al[4], bh[2], bl[2];
#pragma unroll
            for (int ms = 0; ms < 4; ms++) {
                const int r   = wm * 64 + ms * 16 + lr;
                const int off = r * LDST + (lh << 3);
                ah[ms] = mkfrag(&tXh[cur][off], &tXh[cur][off + 16]);
                al[ms] = mkfrag(&tXl[cur][off], &tXl[cur][off + 16]);
            }
            // B frags: lane = column; halves [8*lh..+7] = K(8lh..), then [16+8lh..]
#pragma unroll
            for (int ns = 0; ns < 2; ns++) {
                const int c   = wn * 32 + ns * 16 + lr;
                const int off = c * LDST + (lh << 3);
                bh[ns] = mkfrag(&tSh[cur][off], &tSh[cur][off + 16]);
                bl[ns] = mkfrag(&tSl[cur][off], &tSl[cur][off + 16]);
            }
            // bf16x3: hi*hi + hi*lo + lo*hi into one f32 accumulator
#pragma unroll
            for (int ms = 0; ms < 4; ms++)
#pragma unroll
                for (int ns = 0; ns < 2; ns++) {
                    acc[ms][ns] = wmma_bf16(ah[ms], bh[ns], acc[ms][ns]);
                    acc[ms][ns] = wmma_bf16(ah[ms], bl[ns], acc[ms][ns]);
                    acc[ms][ns] = wmma_bf16(al[ms], bh[ns], acc[ms][ns]);
                }

            // Next buffer's async group was issued at the top of this iter;
            // wait for it, then barrier (also protects buffer recycling).
            if (ks + 1 < KSTEPS) wait_async0();
            __syncthreads();
        }

        // Epilogue: dist = max(0, x2 + s2 - 2*dot); online logsumexp of s*dist.
        const float s20 = s2[nb + wn * 32 + lr];
        const float s21 = s2[nb + wn * 32 + 16 + lr];
#pragma unroll
        for (int ms = 0; ms < 4; ms++) {
#pragma unroll
            for (int r8 = 0; r8 < 8; r8++) {
                const int row = wm * 64 + ms * 16 + (lh << 3) + r8;
                const float xv = sx2[row];
                const float d0 = fmaxf(0.f, xv + s20 - 2.f * acc[ms][0][r8]);
                const float d1 = fmaxf(0.f, xv + s21 - 2.f * acc[ms][1][r8]);
                const float e0 = s * d0, e1 = s * d1;
                const float mi = rm[ms][r8];
                const float nm = fmaxf(mi, fmaxf(e0, e1));
                rl[ms][r8] = rl[ms][r8] * __expf(mi - nm) + __expf(e0 - nm) + __expf(e1 - nm);
                rm[ms][r8] = nm;
            }
        }
    }

    // Merge per-lane (m,l) across the block into per-row partials via LDS.
    __syncthreads();
#pragma unroll
    for (int ms = 0; ms < 4; ms++)
#pragma unroll
        for (int r8 = 0; r8 < 8; r8++) {
            const int row = wm * 64 + ms * 16 + (lh << 3) + r8;
            atomicMin(&rowMu[row], __float_as_uint(rm[ms][r8]));  // all values <= 0
        }
    __syncthreads();
#pragma unroll
    for (int ms = 0; ms < 4; ms++)
#pragma unroll
        for (int r8 = 0; r8 < 8; r8++) {
            const int row = wm * 64 + ms * 16 + (lh << 3) + r8;
            const float M = __uint_as_float(rowMu[row]);
            atomicAdd(&rowL[row], rl[ms][r8] * __expf(rm[ms][r8] - M));  // ds_add_f32
        }
    __syncthreads();
    if (tid < BM) {
        partM[(size_t)(m0 + tid) * NSPLIT + split] = __uint_as_float(rowMu[tid]);
        partL[(size_t)(m0 + tid) * NSPLIT + split] = rowL[tid];
    }
}

// ---------------------------------------------------------------------------
// Kernel 3: combine split partials -> out[i] = (M + log L - log N) / s
// ---------------------------------------------------------------------------
__global__ __launch_bounds__(256) void kde_finalize(const float* __restrict__ partM,
                                                    const float* __restrict__ partL,
                                                    const float* __restrict__ gptr,
                                                    float* __restrict__ out) {
    const int i = blockIdx.x * blockDim.x + threadIdx.x;
    if (i >= Bdim) return;
    const float s = -gptr[0];
    float M = -__builtin_inff();
    for (int j = 0; j < NSPLIT; j++) M = fmaxf(M, partM[(size_t)i * NSPLIT + j]);
    float L = 0.f;
    for (int j = 0; j < NSPLIT; j++)
        L += partL[(size_t)i * NSPLIT + j] * __expf(partM[(size_t)i * NSPLIT + j] - M);
    const float logN = 9.70406052784f;  // log(16384)
    out[i] = (M + __logf(L) - logN) / s;
}

extern "C" void kernel_launch(void* const* d_in, const int* in_sizes, int n_in,
                              void* d_out, int out_size, void* d_ws, size_t ws_size,
                              hipStream_t stream) {
    (void)in_sizes; (void)n_in; (void)out_size; (void)ws_size;
    const float* X   = (const float*)d_in[0];
    const float* svs = (const float*)d_in[1];
    const float* g   = (const float*)d_in[2];
    float* out = (float*)d_out;

    char* ws = (char*)d_ws;
    size_t off = 0;
    unsigned short* Xhi = (unsigned short*)(ws + off); off += (size_t)Bdim * Ddim * 2;
    unsigned short* Xlo = (unsigned short*)(ws + off); off += (size_t)Bdim * Ddim * 2;
    unsigned short* Shi = (unsigned short*)(ws + off); off += (size_t)Ndim * Ddim * 2;
    unsigned short* Slo = (unsigned short*)(ws + off); off += (size_t)Ndim * Ddim * 2;
    float* x2    = (float*)(ws + off); off += (size_t)Bdim * 4;
    float* s2    = (float*)(ws + off); off += (size_t)Ndim * 4;
    float* partM = (float*)(ws + off); off += (size_t)Bdim * NSPLIT * 4;
    float* partL = (float*)(ws + off); off += (size_t)Bdim * NSPLIT * 4;

    kde_split<<<Bdim, 256, 0, stream>>>(X, Xhi, Xlo, x2);
    kde_split<<<Ndim, 256, 0, stream>>>(svs, Shi, Slo, s2);
    kde_gemm_lse<<<dim3(Bdim / BM, NSPLIT), 256, 0, stream>>>(
        Xhi, Xlo, Shi, Slo, x2, s2, g, partM, partL);
    kde_finalize<<<(Bdim + 255) / 256, 256, 0, stream>>>(partM, partL, g, out);
}
